// DiscreteVAE_6373731467566
// MI455X (gfx1250) — compile-verified, matrix operand order checked by reading, not run on hardware
//
#include <hip/hip_runtime.h>
#include <hip/hip_bf16.h>

typedef __attribute__((ext_vector_type(16))) __bf16 v16bf;
typedef __attribute__((ext_vector_type(2)))  __bf16 bf2v;
typedef __attribute__((ext_vector_type(8)))  float  v8f;

#define BM 128
#define BN 128
#define BK 32
#define TPB 256

// ---------------------------------------------------------------------------
// Tensor Data Mover support (gfx1250). Falls back to register staging when the
// builtin is unavailable (also covers the host-side compilation pass).
// ---------------------------------------------------------------------------
#if defined(__AMDGCN__) && __has_builtin(__builtin_amdgcn_tensor_load_to_lds) && \
    __has_builtin(__builtin_amdgcn_s_wait_tensorcnt)
#define USE_TDM 1
#else
#define USE_TDM 0
#endif

#if USE_TDM
typedef __attribute__((ext_vector_type(4))) unsigned int u32x4;
typedef __attribute__((ext_vector_type(8))) int          i32x8;
typedef __attribute__((ext_vector_type(4))) int          i32x4;

// 2-D bf16 tile load: tile_dim0 = 32 elements (64B rows), tile_dim1 = 128 rows,
// LDS pad: +2 DWORDs every 16 DWORDs -> 72B row pitch == [BK+4] __bf16 rows.
// remk/remm: remaining tensor extent from the tile origin (OOB reads -> 0).
__device__ __forceinline__ void tdm_load_tile_2d(unsigned lds_byte_addr,
                                                 const void* gsrc,
                                                 int remk, int remm,
                                                 int stride_elems) {
  unsigned long long ga = (unsigned long long)(uintptr_t)gsrc;
  unsigned rk = remk < 0 ? 0u : (unsigned)remk;
  unsigned rm = remm < 0 ? 0u : (unsigned)remm;
  u32x4 g0;
  g0[0] = 1u;                                   // D# count=1 (user descriptor)
  g0[1] = lds_byte_addr;                        // LDS dest (byte offset)
  g0[2] = (unsigned)ga;                         // global_addr[31:0]
  g0[3] = (unsigned)((ga >> 32) & 0x01FFFFFFull) | (2u << 30);  // addr[56:32], type=2
  i32x8 g1;
  g1[0] = (int)((1u << 16) | (1u << 20) | (3u << 22) | (1u << 25));
          // data_size=2B | pad_enable | pad_interval=16DW | pad_amount=2DW
  g1[1] = (int)((rk & 0xFFFFu) << 16);                         // tensor_dim0 lo
  g1[2] = (int)((rk >> 16) | ((rm & 0xFFFFu) << 16));          // dim0 hi, dim1 lo
  g1[3] = (int)((rm >> 16) | (32u << 16));                     // dim1 hi, tile_dim0=32
  g1[4] = 128;                                                 // tile_dim1=128
  g1[5] = stride_elems;                                        // dim0_stride lo32
  g1[6] = 0;
  g1[7] = 0;
  i32x4 g2 = {0, 0, 0, 0};
  i32x4 g3 = {0, 0, 0, 0};
#if __has_include(<hip/amd_detail/amd_gfx1250_TDM.h>)
  i32x8 gz = {0, 0, 0, 0, 0, 0, 0, 0};
  __builtin_amdgcn_tensor_load_to_lds(g0, g1, g2, g3, gz, 0);
#else
  __builtin_amdgcn_tensor_load_to_lds(g0, g1, g2, g3, 0);
#endif
}
__device__ __forceinline__ void tdm_wait0() { __builtin_amdgcn_s_wait_tensorcnt(0); }
#endif

// ---------------------------------------------------------------------------
// Generic 1-D conv as implicit-GEMM with WMMA bf16 (f32 accumulate).
//   M = Cout, N = B*Tout, K = Cin*KS.  128x128 tile, K-chunk 32, double-
//   buffered LDS; A-tile streamed by the TDM from pre-converted bf16 weights,
//   B-tile (im2col gather) register-staged with clamped unconditional loads.
// ---------------------------------------------------------------------------
struct ConvArgs {
  const float*  __restrict__ x;
  const float*  __restrict__ w;      // f32 weights (fallback path)
  const __bf16* __restrict__ wbf;    // bf16 weights [M][K] (TDM path)
  const float*  __restrict__ bias;
  const float*  __restrict__ resid;  // nullptr if none
  float* __restrict__ y;
  int Bn, Cin, Tin, Cout, Tout, stride, pad, relu;
};

template <int KS, int UPS>
__global__ __launch_bounds__(TPB) void conv1d_wmma(ConvArgs a) {
  __shared__ __bf16 As[2][BM][BK + 4];   // [m][k], 72B row pitch (TDM pad)
  __shared__ __bf16 Bs[2][BN][BK + 4];   // [n][k]

  const int tid  = threadIdx.x;
  const int lane = tid & 31;
  const int wave = tid >> 5;
  const int l15  = lane & 15;
  const int half = lane >> 4;

  const int Ktot  = a.Cin * KS;
  const int n0    = blockIdx.x * BN;
  const int m0    = blockIdx.y * BM;
  const int mbase = wave * 16;

  // B staging: lane -> fixed output position n, 16 consecutive k's.
  const int nB   = tid & 127;
  const int kB   = (tid >> 7) * 16;
  const int gnB  = n0 + nB;
  const int bB   = gnB / a.Tout;
  const int tB   = gnB - bB * a.Tout;
  const int lim  = UPS ? 2 * a.Tin : a.Tin;
  const float* __restrict__ xrow0 = a.x + (size_t)bB * a.Cin * a.Tin;

  v8f acc[8];
#pragma unroll
  for (int j = 0; j < 8; ++j)
#pragma unroll
    for (int r = 0; r < 8; ++r) acc[j][r] = 0.0f;

  const int nk = (Ktot + BK - 1) / BK;

  float breg[16];
  auto loadB = [&](int k0) {
#pragma unroll
    for (int e = 0; e < 16; ++e) {
      int gk  = k0 + kB + e;
      int gkc = gk < Ktot ? gk : Ktot - 1;
      int ci, kk;
      if (KS == 1) { ci = gkc; kk = 0; }
      else         { ci = gkc / 3; kk = gkc - ci * 3; }
      int ti  = (UPS ? tB : tB * a.stride) + kk - a.pad;
      bool ok = (gk < Ktot) & ((unsigned)ti < (unsigned)lim);
      int tic = ti < 0 ? 0 : (ti >= lim ? lim - 1 : ti);
      int xi  = UPS ? (tic >> 1) : tic;
      float v = xrow0[(size_t)ci * a.Tin + xi];   // clamped, unconditional
      breg[e] = ok ? v : 0.0f;
    }
  };
  auto storeB = [&](int buf) {
#pragma unroll
    for (int e = 0; e < 8; ++e) {
      bf2v p;
      p.x = (__bf16)breg[2 * e];
      p.y = (__bf16)breg[2 * e + 1];
      *(bf2v*)&Bs[buf][nB][kB + 2 * e] = p;
    }
  };

#if USE_TDM
  auto issueA = [&](int buf, int k0) {
    if (wave == 0) {
      tdm_load_tile_2d((unsigned)(uintptr_t)&As[buf][0][0],
                       a.wbf + (size_t)m0 * Ktot + k0,
                       Ktot - k0, a.Cout - m0, Ktot);
    }
  };
#else
  float4 areg[4];
  auto loadA = [&](int k0) {
#pragma unroll
    for (int q = 0; q < 4; ++q) {
      int fi = tid + q * TPB;
      int m  = fi >> 3;
      int kq = (fi & 7) * 4;
      int gm = m0 + m, gk = k0 + kq;
      if (gm < a.Cout && gk + 3 < Ktot) {
        areg[q] = *(const float4*)&a.w[(size_t)gm * Ktot + gk];
      } else {
        areg[q].x = areg[q].y = areg[q].z = areg[q].w = 0.0f;
      }
    }
  };
  auto storeA = [&](int buf) {
#pragma unroll
    for (int q = 0; q < 4; ++q) {
      int fi = tid + q * TPB;
      int m  = fi >> 3;
      int kq = (fi & 7) * 4;
      bf2v p0, p1;
      p0.x = (__bf16)areg[q].x; p0.y = (__bf16)areg[q].y;
      p1.x = (__bf16)areg[q].z; p1.y = (__bf16)areg[q].w;
      *(bf2v*)&As[buf][m][kq]     = p0;
      *(bf2v*)&As[buf][m][kq + 2] = p1;
    }
  };
#endif

  // prologue: stage chunk 0 into buffer 0
#if USE_TDM
  issueA(0, 0);
#else
  loadA(0);
#endif
  loadB(0);
  storeB(0);
#if USE_TDM
  if (wave == 0) tdm_wait0();
#else
  storeA(0);
#endif
  __syncthreads();

  for (int kb = 0; kb < nk; ++kb) {
    const int cur = kb & 1, nxt = cur ^ 1;
    const bool more = (kb + 1) < nk;
    if (more) {                       // stream next chunk while computing
#if USE_TDM
      issueA(nxt, (kb + 1) * BK);
#else
      loadA((kb + 1) * BK);
#endif
      loadB((kb + 1) * BK);
    }

    // A fragment (ISA 7.12.2): lane holds row m=lane&15; K-half by lane>>4
    v16bf af;
#pragma unroll
    for (int i = 0; i < 16; ++i) {
      int vg = i >> 1, lo = i & 1;
      int k = (vg < 4 ? 2 * vg : 16 + 2 * (vg - 4)) + 8 * half + lo;
      af[i] = As[cur][mbase + l15][k];
    }
#pragma unroll
    for (int j = 0; j < 8; ++j) {
      v16bf bfr;
#pragma unroll
      for (int i = 0; i < 16; ++i) {
        int vg = i >> 1, lo = i & 1;
        int k = (vg < 4 ? 2 * vg : 16 + 2 * (vg - 4)) + 8 * half + lo;
        bfr[i] = Bs[cur][j * 16 + l15][k];
      }
      acc[j] = __builtin_amdgcn_wmma_f32_16x16x32_bf16(
          false, af, false, bfr, (short)0, acc[j], false, false);
    }

    if (more) {
      storeB(nxt);
#if !USE_TDM
      storeA(nxt);
#endif
    }
#if USE_TDM
    if (more && wave == 0) tdm_wait0();
#endif
    __syncthreads();
  }

  // epilogue: D layout -> n = lane&15, m = r + 8*(lane>>4)
#pragma unroll
  for (int j = 0; j < 8; ++j) {
    int gn = n0 + j * 16 + l15;
    int b = gn / a.Tout, t = gn - b * a.Tout;
#pragma unroll
    for (int r = 0; r < 8; ++r) {
      int gm = m0 + mbase + r + 8 * half;
      if (gm < a.Cout) {
        float v = acc[j][r] + a.bias[gm];
        size_t oi = ((size_t)(b * a.Cout + gm)) * a.Tout + t;
        if (a.resid) v += a.resid[oi];
        if (a.relu) v = v > 0.0f ? v : 0.0f;
        a.y[oi] = v;
      }
    }
  }
}

// ---------------------------------------------------------------------------
// VQ codebook argmin: dist = |e|^2 - 2*z.e  (|z|^2 is row-constant).
// GEMM M=8192 flat rows, N=8192 codes, K=512 over pre-transposed bf16
// operands zT/eT [8192][512]; both tiles streamed by the TDM.
// ---------------------------------------------------------------------------
#define VQ_M  8192
#define VQ_N  8192
#define VQ_K  512
#define VQ_L  256
#define VQ_CB 512

__device__ __forceinline__ unsigned int f2ord(float f) {
  unsigned int u = __float_as_uint(f);
  return (u & 0x80000000u) ? ~u : (u | 0x80000000u);
}
__device__ __forceinline__ unsigned long long shfl_xor_u64(unsigned long long v, int mask) {
  unsigned int lo = (unsigned int)v, hi = (unsigned int)(v >> 32);
  lo = (unsigned int)__shfl_xor((int)lo, mask, 32);
  hi = (unsigned int)__shfl_xor((int)hi, mask, 32);
  return ((unsigned long long)hi << 32) | lo;
}

__global__ __launch_bounds__(TPB) void vq_argmin_wmma(
    const __bf16* __restrict__ zT, const __bf16* __restrict__ eT,
    const float* __restrict__ enorm, unsigned long long* __restrict__ amin) {
  __shared__ __bf16 As[2][BM][BK + 4];
  __shared__ __bf16 Bs[2][BN][BK + 4];

  const int tid  = threadIdx.x;
  const int lane = tid & 31;
  const int wave = tid >> 5;
  const int l15  = lane & 15;
  const int half = lane >> 4;
  const int n0    = blockIdx.x * BN;
  const int m0    = blockIdx.y * BM;
  const int mbase = wave * 16;

  v8f acc[8];
#pragma unroll
  for (int j = 0; j < 8; ++j)
#pragma unroll
    for (int r = 0; r < 8; ++r) acc[j][r] = 0.0f;

#if USE_TDM
  auto issueAB = [&](int buf, int k0) {
    if (wave == 0) {
      tdm_load_tile_2d((unsigned)(uintptr_t)&As[buf][0][0],
                       zT + (size_t)m0 * VQ_K + k0, VQ_K - k0, VQ_M - m0, VQ_K);
      tdm_load_tile_2d((unsigned)(uintptr_t)&Bs[buf][0][0],
                       eT + (size_t)n0 * VQ_K + k0, VQ_K - k0, VQ_N - n0, VQ_K);
    }
  };
  issueAB(0, 0);
  if (wave == 0) tdm_wait0();
  __syncthreads();
#else
  const int rS = tid & 127;
  const int kS = (tid >> 7) * 16;
  float aregv[16], bregv[16];
  auto loadAB = [&](int k0) {
#pragma unroll
    for (int e = 0; e < 16; ++e) {
      int gk = k0 + kS + e;
      aregv[e] = (float)zT[(size_t)(m0 + rS) * VQ_K + gk];
      bregv[e] = (float)eT[(size_t)(n0 + rS) * VQ_K + gk];
    }
  };
  auto storeAB = [&](int buf) {
#pragma unroll
    for (int e = 0; e < 8; ++e) {
      bf2v pa, pb;
      pa.x = (__bf16)aregv[2 * e]; pa.y = (__bf16)aregv[2 * e + 1];
      pb.x = (__bf16)bregv[2 * e]; pb.y = (__bf16)bregv[2 * e + 1];
      *(bf2v*)&As[buf][rS][kS + 2 * e] = pa;
      *(bf2v*)&Bs[buf][rS][kS + 2 * e] = pb;
    }
  };
  loadAB(0);
  storeAB(0);
  __syncthreads();
#endif

  const int nk = VQ_K / BK;
  for (int kb = 0; kb < nk; ++kb) {
    const int cur = kb & 1, nxt = cur ^ 1;
    const bool more = (kb + 1) < nk;
#if USE_TDM
    if (more) issueAB(nxt, (kb + 1) * BK);
#else
    if (more) loadAB((kb + 1) * BK);
#endif

    v16bf af;
#pragma unroll
    for (int i = 0; i < 16; ++i) {
      int vg = i >> 1, lo = i & 1;
      int k = (vg < 4 ? 2 * vg : 16 + 2 * (vg - 4)) + 8 * half + lo;
      af[i] = As[cur][mbase + l15][k];
    }
#pragma unroll
    for (int j = 0; j < 8; ++j) {
      v16bf bfr;
#pragma unroll
      for (int i = 0; i < 16; ++i) {
        int vg = i >> 1, lo = i & 1;
        int k = (vg < 4 ? 2 * vg : 16 + 2 * (vg - 4)) + 8 * half + lo;
        bfr[i] = Bs[cur][j * 16 + l15][k];
      }
      acc[j] = __builtin_amdgcn_wmma_f32_16x16x32_bf16(
          false, af, false, bfr, (short)0, acc[j], false, false);
    }

#if USE_TDM
    if (more && wave == 0) tdm_wait0();
#else
    if (more) storeAB(nxt);
#endif
    __syncthreads();
  }

  // per-row argmin over this block's 128 columns, then global merge
#pragma unroll
  for (int r = 0; r < 8; ++r) {
    unsigned long long best = ~0ull;
#pragma unroll
    for (int j = 0; j < 8; ++j) {
      int gn = n0 + j * 16 + l15;
      float dist = enorm[gn] - 2.0f * acc[j][r];
      unsigned long long p =
          ((unsigned long long)f2ord(dist) << 32) | (unsigned int)gn;
      best = p < best ? p : best;
    }
#pragma unroll
    for (int mask = 1; mask < 16; mask <<= 1) {
      unsigned long long o = shfl_xor_u64(best, mask);
      best = o < best ? o : best;
    }
    if (l15 == 0) {
      int gm = m0 + mbase + r + 8 * half;
      atomicMin(&amin[gm], best);
    }
  }
}

// ---------------------------------------------------------------------------
// Small helper kernels
// ---------------------------------------------------------------------------
__global__ void cvt_bf16_kernel(const float* __restrict__ s, __bf16* __restrict__ d, int n) {
  int i = blockIdx.x * 256 + threadIdx.x;
  if (i < n) d[i] = (__bf16)s[i];
}

// zT[m][c] (bf16, [8192][512]) from z [32][512][256] f32; m = b*256 + l
__global__ void transpose_z_kernel(const float* __restrict__ z, __bf16* __restrict__ zT) {
  int i = blockIdx.x * 256 + threadIdx.x;    // over 8192*512
  int c = i & 511, m = i >> 9;
  int b = m >> 8, l = m & 255;
  zT[i] = (__bf16)z[(((size_t)(b * VQ_CB + c)) << 8) + l];
}

// eT[n][k] (bf16, [8192][512]) from embed [512][8192] f32
__global__ void transpose_e_kernel(const float* __restrict__ e, __bf16* __restrict__ eT) {
  int i = blockIdx.x * 256 + threadIdx.x;    // over 8192*512
  int k = i & 511, n = i >> 9;
  eT[i] = (__bf16)e[((size_t)k << 13) + n];
}

__global__ void enorm_kernel(const float* __restrict__ embed, float* __restrict__ enorm) {
  int n = blockIdx.x * blockDim.x + threadIdx.x;
  if (n < VQ_N) {
    float s = 0.0f;
    for (int k = 0; k < VQ_K; ++k) {
      float e = embed[(size_t)k * VQ_N + n];
      s += e * e;
    }
    enorm[n] = s;
  }
}

__global__ void vq_init_kernel(unsigned long long* __restrict__ amin, float* __restrict__ diffsum) {
  int i = blockIdx.x * blockDim.x + threadIdx.x;
  if (i < VQ_M) amin[i] = ~0ull;
  if (i == 0) diffsum[0] = 0.0f;
}

__global__ __launch_bounds__(256) void vq_gather_diff(
    const float* __restrict__ embed, const unsigned long long* __restrict__ amin,
    const float* __restrict__ z, float* __restrict__ q, float* __restrict__ diffsum) {
  __shared__ float red[256];
  const int tot = 32 * VQ_CB * VQ_L;
  int i = blockIdx.x * 256 + threadIdx.x;
  float d = 0.0f;
  if (i < tot) {
    int l = i & 255;
    int bc = i >> 8;
    int c = bc & 511;
    int b = bc >> 9;
    unsigned int idx = (unsigned int)(amin[b * VQ_L + l] & 0xFFFFFFFFull);
    float qv = embed[(size_t)c * VQ_N + idx];
    q[i] = qv;               // straight-through: forward value == codebook entry
    float t = qv - z[i];
    d = t * t;
  }
  red[threadIdx.x] = d;
  __syncthreads();
  for (int s = 128; s > 0; s >>= 1) {
    if (threadIdx.x < s) red[threadIdx.x] += red[threadIdx.x + s];
    __syncthreads();
  }
  if (threadIdx.x == 0) atomicAdd(diffsum, red[0]);
}

__global__ void finalize_kernel(const float* __restrict__ mel, const float* __restrict__ outb,
                                const float* __restrict__ diffsum, float* __restrict__ dout) {
  const int N = 32 * 80 * 1024;
  int i = blockIdx.x * blockDim.x + threadIdx.x;
  if (i < N) {
    float m = mel[i], o = outb[i];
    float r = m - o;
    dout[i] = r * r;        // recon
    dout[N + 1 + i] = o;    // out
  }
  if (i == 0) dout[N] = diffsum[0] * (1.0f / 4194304.0f);  // diff = mean
}

// ---------------------------------------------------------------------------
// Host: pipeline
// ---------------------------------------------------------------------------
extern "C" void kernel_launch(void* const* d_in, const int* in_sizes, int n_in,
                              void* d_out, int out_size, void* d_ws, size_t ws_size,
                              hipStream_t stream) {
  (void)in_sizes; (void)n_in; (void)out_size; (void)ws_size;

  const float* mel   = (const float*)d_in[0];
  const float* embed = (const float*)d_in[1];
  int ip = 2;
  auto nextp = [&]() { return (const float*)d_in[ip++]; };
  const float *e0w = nextp(), *e0b = nextp();
  const float *e1w = nextp(), *e1b = nextp();
  const float *erw[3][3], *erb[3][3];
  for (int r = 0; r < 3; ++r)
    for (int j = 0; j < 3; ++j) { erw[r][j] = nextp(); erb[r][j] = nextp(); }
  const float *eow = nextp(), *eob = nextp();
  const float *diw = nextp(), *dib = nextp();
  const float *drw[3][3], *drb[3][3];
  for (int r = 0; r < 3; ++r)
    for (int j = 0; j < 3; ++j) { drw[r][j] = nextp(); drb[r][j] = nextp(); }
  const float *u0w = nextp(), *u0b = nextp();
  const float *u1w = nextp(), *u1b = nextp();
  const float *dow = nextp(), *dob = nextp();

  // workspace carve-up (float units)
  float* ws = (float*)d_ws;
  size_t off = 0;
  float* buf0 = ws + off; off += 16777216;   // up to [32,1024,512]
  float* buf1 = ws + off; off += 16777216;
  float* buf2 = ws + off; off += 16777216;
  float* zbuf = ws + off; off += 4194304;    // [32,512,256] encoder logits (z)
  float* qbuf = ws + off; off += 4194304;    // [32,512,256] quantized
  float* outb = ws + off; off += 2621440;    // [32,80,1024]
  float* enorm = ws + off; off += 8192;
  unsigned long long* amin = (unsigned long long*)(ws + off); off += 16384;
  float* diffsum = ws + off; off += 4;
  __bf16* zT = (__bf16*)(ws + off); off += 2097152;   // [8192][512] bf16
  __bf16* eT = (__bf16*)(ws + off); off += 2097152;   // [8192][512] bf16
  __bf16* wpool = (__bf16*)(ws + off);                // bf16 weight pool (~52M elems)
  size_t wcur = 0;

  auto tobf = [&](const float* src, size_t n) -> const __bf16* {
    __bf16* dst = wpool + wcur;
    wcur += (n + 7) & ~(size_t)7;
    cvt_bf16_kernel<<<dim3((unsigned)((n + 255) / 256)), dim3(256), 0, stream>>>(
        src, dst, (int)n);
    return dst;
  };

  auto conv = [&](const float* x, const float* w, const float* b, const float* resid,
                  float* y, int Cin, int Tin, int Cout, int Tout,
                  int KS, int stride, int pad, int relu, int ups) {
    const __bf16* wbf = tobf(w, (size_t)Cout * Cin * KS);
    ConvArgs a{x, w, wbf, b, resid, y, 32, Cin, Tin, Cout, Tout, stride, pad, relu};
    dim3 g((unsigned)((32 * Tout) / BN), (unsigned)((Cout + BM - 1) / BM));
    if (KS == 1)      conv1d_wmma<1, 0><<<g, dim3(TPB), 0, stream>>>(a);
    else if (!ups)    conv1d_wmma<3, 0><<<g, dim3(TPB), 0, stream>>>(a);
    else              conv1d_wmma<3, 1><<<g, dim3(TPB), 0, stream>>>(a);
  };

  // ---- encoder ----
  conv(mel,  e0w, e0b, nullptr, buf0,  80, 1024,  512,  512, 3, 2, 1, 1, 0);  // relu
  conv(buf0, e1w, e1b, nullptr, buf1, 512,  512, 1024,  256, 3, 2, 1, 1, 0);  // relu
  for (int r = 0; r < 3; ++r) {
    conv(buf1, erw[r][0], erb[r][0], nullptr, buf0, 1024, 256, 1024, 256, 3, 1, 1, 1, 0);
    conv(buf0, erw[r][1], erb[r][1], nullptr, buf2, 1024, 256, 1024, 256, 3, 1, 1, 1, 0);
    conv(buf2, erw[r][2], erb[r][2], buf1,    buf1, 1024, 256, 1024, 256, 1, 1, 0, 0, 0);
  }
  conv(buf1, eow, eob, nullptr, zbuf, 1024, 256, 512, 256, 1, 1, 0, 0, 0);

  // ---- vector quantize ----
  enorm_kernel<<<dim3(VQ_N / 256), dim3(256), 0, stream>>>(embed, enorm);
  vq_init_kernel<<<dim3(VQ_M / 256), dim3(256), 0, stream>>>(amin, diffsum);
  transpose_e_kernel<<<dim3((VQ_N * VQ_K) / 256), dim3(256), 0, stream>>>(embed, eT);
  transpose_z_kernel<<<dim3((VQ_M * VQ_K) / 256), dim3(256), 0, stream>>>(zbuf, zT);
  vq_argmin_wmma<<<dim3(VQ_N / BN, VQ_M / BM), dim3(TPB), 0, stream>>>(zT, eT, enorm, amin);
  vq_gather_diff<<<dim3((32 * VQ_CB * VQ_L) / 256), dim3(256), 0, stream>>>(
      embed, amin, zbuf, qbuf, diffsum);

  // ---- decoder ----
  conv(qbuf, diw, dib, nullptr, buf1, 512, 256, 1024, 256, 1, 1, 0, 0, 0);
  for (int r = 0; r < 3; ++r) {
    conv(buf1, drw[r][0], drb[r][0], nullptr, buf0, 1024, 256, 1024, 256, 3, 1, 1, 1, 0);
    conv(buf0, drw[r][1], drb[r][1], nullptr, buf2, 1024, 256, 1024, 256, 3, 1, 1, 1, 0);
    conv(buf2, drw[r][2], drb[r][2], buf1,    buf1, 1024, 256, 1024, 256, 1, 1, 0, 0, 0);
  }
  conv(buf1, u0w, u0b, nullptr, buf0, 1024, 256, 1024,  512, 3, 1, 1, 1, 1);  // up x2 + relu
  conv(buf0, u1w, u1b, nullptr, buf2, 1024, 512,  512, 1024, 3, 1, 1, 1, 1);  // up x2 + relu
  conv(buf2, dow, dob, nullptr, outb, 512, 1024,   80, 1024, 1, 1, 0, 0, 0);

  // ---- outputs: (recon, diff, out) ----
  finalize_kernel<<<dim3((32 * 80 * 1024 + 255) / 256), dim3(256), 0, stream>>>(
      mel, outb, diffsum, (float*)d_out);
}
